// VanillaRNN_4346506903784
// MI455X (gfx1250) — matrix operand order repaired and checked by reading, not compiled, not run on hardware
//
#include <hip/hip_runtime.h>
#include <math.h>

typedef __bf16 bf16;
typedef __attribute__((ext_vector_type(16))) __bf16 v16bf;
typedef __attribute__((ext_vector_type(8)))  float  v8f;

static constexpr int RB = 64;    // batch
static constexpr int RH = 512;   // d_input
static constexpr int RL = 512;   // sequence length
static constexpr int RS = 1024;  // d_state
static constexpr int LDSP = 1032; // padded LDS row stride (bf16 elems): +16B kills bank conflicts

// Direct global->LDS async copy (CDNA5, ASYNCcnt-tracked), with safe fallback.
#if defined(__AMDGCN__) && __has_builtin(__builtin_amdgcn_global_load_async_to_lds_b128)
#define HAVE_ASYNC_LDS 1
// Exact parameter types per hipcc diagnostic: int32x4 (vector_size(16)) pointers,
// global (AS1, printed "__device__") source and LDS (AS3, "__shared__") destination.
typedef int vsi4 __attribute__((__vector_size__(16)));
typedef vsi4 __attribute__((address_space(1))) g_vsi4;
typedef vsi4 __attribute__((address_space(3))) l_vsi4;
#endif

// ---------------- WMMA fragment loaders (layouts per CDNA5 ISA) ----------------

// A-matrix 16x32 bf16: lane l -> row m = l&15, chunk base c = l>>4:
//   halves 0..7  = K[kb + c*8 .. +7],  halves 8..15 = K[kb + c*8 + 16 .. +7]
__device__ __forceinline__ v16bf ldA_global(const bf16* base, long row, long ld, int kb, int c) {
  const bf16* p = base + row * ld + kb + c * 8;
  v16bf a;
#pragma unroll
  for (int j = 0; j < 8; ++j) { a[j] = p[j]; a[8 + j] = p[16 + j]; }
  return a;
}

__device__ __forceinline__ v16bf ldA_lds(const bf16* shbase, int m, int kb, int c) {
  const bf16* p = shbase + m * LDSP + kb + c * 8;
  v16bf a;
#pragma unroll
  for (int j = 0; j < 8; ++j) { a[j] = p[j]; a[8 + j] = p[16 + j]; }
  return a;
}

// B-matrix 32x16 bf16 (K x N): lane l -> col n = l&15, c = l>>4:
//   halves 0..7 = K[kb + c*8 .. +7], halves 8..15 = K[kb + c*8 + 16 .. +7] (same as A)
__device__ __forceinline__ v16bf ldB_row(const bf16* wbase, long nrow, long ld, int kb, int c) {
  const bf16* p = wbase + nrow * ld + kb + c * 8;
  v16bf b;
#pragma unroll
  for (int j = 0; j < 8; ++j) { b[j] = p[j]; b[8 + j] = p[16 + j]; }
  return b;
}

// ---- stage 16 rows x RS bf16 of h into padded LDS (16B chunks, no divergence) ----
__device__ __forceinline__ void stage_h_to_lds(const bf16* __restrict__ src,
                                               bf16* sh, int tid) {
#if defined(HAVE_ASYNC_LDS)
  for (int idx = tid; idx < 16 * (RS / 8); idx += 256) {
    const int r  = idx >> 7;          // RS/8 = 128 chunks per row
    const int kc = (idx & 127) * 8;
    __builtin_amdgcn_global_load_async_to_lds_b128(
        (g_vsi4*)(src + (long)r * RS + kc),
        (l_vsi4*)(sh + r * LDSP + kc),
        0, 0);
  }
#if __has_builtin(__builtin_amdgcn_s_wait_asynccnt)
  __builtin_amdgcn_s_wait_asynccnt(0);
#else
  asm volatile("s_wait_asynccnt 0x0" ::: "memory");
#endif
#else
  for (int idx = tid; idx < 16 * (RS / 8); idx += 256) {
    const int r  = idx >> 7;
    const int kc = (idx & 127) * 8;
#pragma unroll
    for (int j = 0; j < 8; ++j) sh[r * LDSP + kc + j] = src[(long)r * RS + kc + j];
  }
#endif
}

// ---------------- Kernel 0a: LDS-tiled transpose  x(B,H,L) f32 -> xbt[(l*B+b)*H + h] bf16 ----
__global__ __launch_bounds__(256) void k_transpose_x(const float* __restrict__ x,
                                                     bf16* __restrict__ xbt) {
  __shared__ float tile[32][33];
  const int b  = blockIdx.z;
  const int ht = blockIdx.y;      // h tile
  const int lt = blockIdx.x;      // l tile
  const int tx = threadIdx.x;     // 0..31
  for (int i = threadIdx.y; i < 32; i += 8) {
    const int h = ht * 32 + i;
    const int l = lt * 32 + tx;
    tile[i][tx] = x[((long)(b * RH + h)) * RL + l];
  }
  __syncthreads();
  for (int i = threadIdx.y; i < 32; i += 8) {
    const int l = lt * 32 + i;
    const int h = ht * 32 + tx;
    xbt[((long)(l * RB + b)) * RH + h] = (bf16)tile[tx][i];
  }
}

// ---------------- Kernel 0b: convert weights to bf16, zero h buffers + barrier counter -------
__global__ __launch_bounds__(256) void k_prep(const float* __restrict__ W_ih,
                                              const float* __restrict__ W_hh,
                                              const float* __restrict__ W_fc,
                                              bf16* __restrict__ w_ihb,
                                              bf16* __restrict__ w_hhb,
                                              bf16* __restrict__ w_fcb,
                                              bf16* __restrict__ hbuf,
                                              int* __restrict__ ctr) {
  const long tid    = (long)blockIdx.x * blockDim.x + threadIdx.x;
  const long stride = (long)gridDim.x * blockDim.x;
  for (long i = tid; i < (long)RS * RH; i += stride) w_ihb[i] = (bf16)W_ih[i];
  for (long i = tid; i < (long)RS * RS; i += stride) w_hhb[i] = (bf16)W_hh[i];
  for (long i = tid; i < (long)RH * RS; i += stride) w_fcb[i] = (bf16)W_fc[i];
  for (long i = tid; i < (long)2 * RB * RS; i += stride) hbuf[i] = (bf16)0.0f;
  if (tid == 0) *ctr = 0;
}

// ---------------- Kernel 1: xa[(l*B+b), s] = xbt @ W_ih^T + b_ih + b_hh  (bf16 WMMA) ---------
__global__ __launch_bounds__(256) void k_gemm_xa(const bf16* __restrict__ xbt,
                                                 const bf16* __restrict__ w_ihb,
                                                 const float* __restrict__ b_ih,
                                                 const float* __restrict__ b_hh,
                                                 float* __restrict__ xa) {
  const int w    = blockIdx.x * 8 + (threadIdx.x >> 5);
  const int lane = threadIdx.x & 31;
  const int ln   = lane & 15;
  const int c    = lane >> 4;
  const int tm   = w >> 6;         // 0..2047  (M = L*B = 32768)
  const int tn   = w & 63;         // 0..63    (N = S = 1024)
  const long row0 = (long)tm * 16;
  const int  col0 = tn * 16;

  const float bias = b_ih[col0 + ln] + b_hh[col0 + ln];
  v8f acc;
#pragma unroll
  for (int v = 0; v < 8; ++v) acc[v] = bias;

  for (int kb = 0; kb < RH; kb += 32) {
    v16bf a = ldA_global(xbt, row0 + ln, RH, kb, c);
    v16bf b = ldB_row(w_ihb, col0 + ln, RH, kb, c);
    acc = __builtin_amdgcn_wmma_f32_16x16x32_bf16(false, a, false, b, (short)0, acc, false, false);
  }
#pragma unroll
  for (int v = 0; v < 8; ++v) {
    const long r = row0 + v + 8 * c;
    xa[r * RS + col0 + ln] = acc[v];
  }
}

// ---------------- Kernel 2: persistent scan: h = tanh(xa_t + h W_hh^T); y_t = h W_fc^T + b_fc
// 32 blocks x 256 threads = 256 waves. Each wave owns one 16x16 tile of h (4 bi x 64 sj).
// Device-wide barrier (monotonic atomic counter) once per step; h double-buffered in global.
__global__ __launch_bounds__(256) void k_rnn_scan(const float* __restrict__ xa,
                                                  const bf16* __restrict__ w_hhb,
                                                  const bf16* __restrict__ w_fcb,
                                                  const float* __restrict__ b_fc,
                                                  bf16* __restrict__ hbuf,
                                                  float* __restrict__ out,
                                                  int* __restrict__ ctr) {
  __shared__ bf16 sh[16 * LDSP];   // 33 KB: 16 rows of h (this block's batch rows), padded

  const int tid  = threadIdx.x;
  const int lane = tid & 31;
  const int wid  = tid >> 5;                  // 0..7
  const int ln   = lane & 15;
  const int c    = lane >> 4;
  const int bi   = blockIdx.x >> 3;           // 0..3 : batch tile (shared by the whole block)
  const int sj   = ((blockIdx.x & 7) << 3) | wid;   // 0..63 : d_state tile
  const bool yact = (blockIdx.x & 7) < 4;           // 16 blocks handle the y projection
  const int oj   = ((blockIdx.x & 3) << 3) | wid;   // 0..31 : output tile (when yact)
  const int nblk = gridDim.x;

  for (int l = 0; l < RL; ++l) {
    const int cur = l & 1, nxt = cur ^ 1;
    const bf16* hc = hbuf + (long)cur * RB * RS;
    bf16*       hn = hbuf + (long)nxt * RB * RS;

    __syncthreads();  // protect sh against previous iteration's readers
    stage_h_to_lds(hc + (long)(bi * 16) * RS, sh, tid);
    __syncthreads();

    // --- h update: acc = xa[l] tile + h @ W_hh^T ; tanh ; store bf16 to hn ---
    v8f acc;
    {
      const float* xrow = xa + ((long)(l * RB + bi * 16)) * RS + sj * 16 + ln;
#pragma unroll
      for (int v = 0; v < 8; ++v) acc[v] = xrow[(long)(v + 8 * c) * RS];
    }
    for (int kb = 0; kb < RS; kb += 32) {
      v16bf a = ldA_lds(sh, ln, kb, c);
      v16bf b = ldB_row(w_hhb, sj * 16 + ln, RS, kb, c);
      acc = __builtin_amdgcn_wmma_f32_16x16x32_bf16(false, a, false, b, (short)0, acc, false, false);
    }
#pragma unroll
    for (int v = 0; v < 8; ++v) acc[v] = tanhf(acc[v]);
#pragma unroll
    for (int v = 0; v < 8; ++v)
      hn[(long)(bi * 16 + v + 8 * c) * RS + sj * 16 + ln] = (bf16)acc[v];

    // --- device-wide barrier: all 256 tiles of h_{l+1} written ---
    __threadfence();
    __syncthreads();
    if (tid == 0) {
      __hip_atomic_fetch_add(ctr, 1, __ATOMIC_ACQ_REL, __HIP_MEMORY_SCOPE_AGENT);
      const int target = nblk * (l + 1);
      while (__hip_atomic_load(ctr, __ATOMIC_ACQUIRE, __HIP_MEMORY_SCOPE_AGENT) < target)
        __builtin_amdgcn_s_sleep(2);
    }
    __syncthreads();
    __threadfence();

    // --- y_t = h_{l+1} @ W_fc^T + b_fc, written flat in (L,B,H) order ---
    if (yact) {
      stage_h_to_lds(hn + (long)(bi * 16) * RS, sh, tid);
      __syncthreads();

      v8f yacc;
      const float bias = b_fc[oj * 16 + ln];
#pragma unroll
      for (int v = 0; v < 8; ++v) yacc[v] = bias;
      for (int kb = 0; kb < RS; kb += 32) {
        v16bf a = ldA_lds(sh, ln, kb, c);
        v16bf b = ldB_row(w_fcb, oj * 16 + ln, RS, kb, c);
        yacc = __builtin_amdgcn_wmma_f32_16x16x32_bf16(false, a, false, b, (short)0, yacc, false, false);
      }
#pragma unroll
      for (int v = 0; v < 8; ++v)
        out[((long)(l * RB + bi * 16 + v + 8 * c)) * RH + oj * 16 + ln] = yacc[v];
    }
    // No second barrier needed: step l+1 writes h[cur], which nobody reads for y;
    // step l+2 writes h[nxt] only after barrier l+1, which y readers of step l join last.
  }
}

// ---------------- host-side launcher ----------------
extern "C" void kernel_launch(void* const* d_in, const int* in_sizes, int n_in,
                              void* d_out, int out_size, void* d_ws, size_t ws_size,
                              hipStream_t stream) {
  (void)in_sizes; (void)n_in; (void)out_size; (void)ws_size;
  const float* x    = (const float*)d_in[0];   // (B,H,L)
  const float* W_ih = (const float*)d_in[1];   // (S,H)
  const float* W_hh = (const float*)d_in[2];   // (S,S)
  const float* b_ih = (const float*)d_in[3];   // (S)
  const float* b_hh = (const float*)d_in[4];   // (S)
  const float* W_fc = (const float*)d_in[5];   // (H,S)
  const float* b_fc = (const float*)d_in[6];   // (H)
  float* out = (float*)d_out;                  // (B,H,L) flat == (L,B,H) flat (raw reshape)

  // workspace layout (all 256B-aligned by construction)
  constexpr size_t XA_B   = (size_t)RL * RB * RS * sizeof(float);  // 128 MB
  constexpr size_t XBT_B  = (size_t)RL * RB * RH * sizeof(bf16);   //  32 MB
  constexpr size_t WIH_B  = (size_t)RS * RH * sizeof(bf16);
  constexpr size_t WHH_B  = (size_t)RS * RS * sizeof(bf16);
  constexpr size_t WFC_B  = (size_t)RH * RS * sizeof(bf16);
  constexpr size_t HBUF_B = (size_t)2 * RB * RS * sizeof(bf16);

  char* wsb = (char*)d_ws;
  float* xa   = (float*)(wsb);
  bf16* xbt   = (bf16*)(wsb + XA_B);
  bf16* w_ihb = (bf16*)(wsb + XA_B + XBT_B);
  bf16* w_hhb = (bf16*)(wsb + XA_B + XBT_B + WIH_B);
  bf16* w_fcb = (bf16*)(wsb + XA_B + XBT_B + WIH_B + WHH_B);
  bf16* hbuf  = (bf16*)(wsb + XA_B + XBT_B + WIH_B + WHH_B + WFC_B);
  int*  ctr   = (int*) (wsb + XA_B + XBT_B + WIH_B + WHH_B + WFC_B + HBUF_B);

  // 0a: transpose+convert x ; 0b: convert weights, zero h buffers + barrier counter
  k_transpose_x<<<dim3(RL / 32, RH / 32, RB), dim3(32, 8), 0, stream>>>(x, xbt);
  k_prep<<<1024, 256, 0, stream>>>(W_ih, W_hh, W_fc, w_ihb, w_hhb, w_fcb, hbuf, ctr);

  // 1: big input-projection GEMM: (L*B/16)*(S/16) = 131072 wave tiles, 8 waves/block
  k_gemm_xa<<<(RL * RB / 16) * (RS / 16) / 8, 256, 0, stream>>>(xbt, w_ihb, b_ih, b_hh, xa);

  // 2: persistent recurrent scan + fused output projection (32 co-resident blocks)
  k_rnn_scan<<<32, 256, 0, stream>>>(xa, w_hhb, w_fcb, b_fc, hbuf, out, ctr);
}
